// DynamicMoE_22265110463279
// MI455X (gfx1250) — matrix operand order repaired and checked by reading, not compiled, not run on hardware
//
#include <hip/hip_runtime.h>
#include <hip/hip_bf16.h>
#include <math.h>

typedef _Float16 v16h __attribute__((ext_vector_type(16)));
typedef _Float16 v8h  __attribute__((ext_vector_type(8)));
typedef _Float16 v4h  __attribute__((ext_vector_type(4)));
typedef float    v8f  __attribute__((ext_vector_type(8)));

#define NTOK 8192
#define DIMD 1024
#define DIMH 4096
#define NEXP 8
#define HCHUNK 1024
#define MTILE 32
#define XSTR (1024 + 8)   // halfs; 2064B row stride -> 4-bank rotation, conflict-free b128
#define HSTR (1024 + 8)

static __device__ __forceinline__ v8f wmma_f16(v16h a, v16h b, v8f c) {
    // D = A(16x32 f16) x B(32x16 f16) + C(16x16 f32)
    return __builtin_amdgcn_wmma_f32_16x16x32_f16(false, a, false, b, (short)0, c,
                                                  false, false);
}

// A-fragment (16x32 f16): lane m=lane&15, kb=8*(lane>>4).
// elems 0..15 = A[m][k0+2*kb .. +15] (16 consecutive K, matches B)
static __device__ __forceinline__ v16h afrag_lds(const _Float16* rowk, int kb) {
    v8h lo = *(const v8h*)(rowk + 2 * kb);
    v8h hi = *(const v8h*)(rowk + 2 * kb + 8);
    return __builtin_shufflevector(lo, hi, 0,1,2,3,4,5,6,7,8,9,10,11,12,13,14,15);
}

// B-fragment (32x16 f16): lane n=lane&15, K group = 16*(lane>>4):
// 16 consecutive K of one weight row.
static __device__ __forceinline__ v16h bfrag(const _Float16* p) {
    v8h lo = *(const v8h*)(p);
    v8h hi = *(const v8h*)(p + 8);
    return __builtin_shufflevector(lo, hi, 0,1,2,3,4,5,6,7,8,9,10,11,12,13,14,15);
}
static __device__ __forceinline__ v16h bfrag(const float* p) {
    float4 f0 = ((const float4*)p)[0];
    float4 f1 = ((const float4*)p)[1];
    float4 f2 = ((const float4*)p)[2];
    float4 f3 = ((const float4*)p)[3];
    v16h r;
    r[0]=(_Float16)f0.x; r[1]=(_Float16)f0.y; r[2]=(_Float16)f0.z; r[3]=(_Float16)f0.w;
    r[4]=(_Float16)f1.x; r[5]=(_Float16)f1.y; r[6]=(_Float16)f1.z; r[7]=(_Float16)f1.w;
    r[8]=(_Float16)f2.x; r[9]=(_Float16)f2.y; r[10]=(_Float16)f2.z; r[11]=(_Float16)f2.w;
    r[12]=(_Float16)f3.x; r[13]=(_Float16)f3.y; r[14]=(_Float16)f3.z; r[15]=(_Float16)f3.w;
    return r;
}

__global__ void moe_zero_counts(int* counts) {
    if (threadIdx.x < NEXP) counts[threadIdx.x] = 0;
}

// fp32 -> f16 weight conversion (grid-stride, float4 in / v4h out)
__global__ void moe_cvt_w(const float* __restrict__ w, _Float16* __restrict__ wh, long n4) {
    long i = (long)blockIdx.x * blockDim.x + threadIdx.x;
    long stride = (long)gridDim.x * blockDim.x;
    for (; i < n4; i += stride) {
        float4 v = ((const float4*)w)[i];
        v4h h = { (_Float16)v.x, (_Float16)v.y, (_Float16)v.z, (_Float16)v.w };
        ((v4h*)wh)[i] = h;
    }
}

// One wave32 per token: logits, softmax, top-2, route to e* = max(i1,i2)
// (reference's jnp.where-overwrite semantics), scale = softmax score of e*.
__global__ void moe_gate(const float* __restrict__ x, const float* __restrict__ gw,
                         const float* __restrict__ gb, int* __restrict__ counts,
                         int* __restrict__ toks, float* __restrict__ scl) {
    int tok = blockIdx.x * (blockDim.x >> 5) + (threadIdx.x >> 5);
    if (tok >= NTOK) return;
    int lane = threadIdx.x & 31;
    const float* xr = x + (size_t)tok * DIMD;
    float acc[NEXP];
#pragma unroll
    for (int e = 0; e < NEXP; ++e) acc[e] = 0.f;
    for (int d = lane; d < DIMD; d += 32) {
        float xv = xr[d];
#pragma unroll
        for (int e = 0; e < NEXP; ++e) acc[e] += xv * gw[e * DIMD + d];
    }
#pragma unroll
    for (int e = 0; e < NEXP; ++e) {
        float v = acc[e];
#pragma unroll
        for (int m = 16; m >= 1; m >>= 1) v += __shfl_xor(v, m, 32);
        acc[e] = v;
    }
    if (lane == 0) {
        float lg[NEXP];
        float mx = -INFINITY;
#pragma unroll
        for (int e = 0; e < NEXP; ++e) { lg[e] = acc[e] + gb[e]; mx = fmaxf(mx, lg[e]); }
        float s = 0.f;
#pragma unroll
        for (int e = 0; e < NEXP; ++e) s += __expf(lg[e] - mx);
        int i1 = 0;
#pragma unroll
        for (int e = 1; e < NEXP; ++e) if (lg[e] > lg[i1]) i1 = e;
        int i2 = (i1 == 0) ? 1 : 0;
#pragma unroll
        for (int e = 0; e < NEXP; ++e) if (e != i1 && lg[e] > lg[i2]) i2 = e;
        int es = i1 > i2 ? i1 : i2;
        float scale = __expf(lg[es] - mx) / s;
        int pos = atomicAdd(&counts[es], 1);
        toks[es * NTOK + pos] = tok;
        scl[es * NTOK + pos] = scale;
    }
}

// One WG (8 waves, 256 thr) per (expert, 32-token tile).
// LDS: Xs (32x1024 f16, padded) + Hs (32x1024 f16, padded) ~ 132 KB.
// Persistent per-wave y accumulators: 32 rows x 128 cols f32 = 128 VGPRs.
// WT = _Float16 (preconverted weights in ws) or float (fallback, cvt in-loop).
template <typename WT>
__global__ __launch_bounds__(256)
void moe_ffn(const float* __restrict__ x,
             const WT* __restrict__ w1, const float* __restrict__ b1,
             const WT* __restrict__ w2, const float* __restrict__ b2,
             const int* __restrict__ counts, const int* __restrict__ toks,
             const float* __restrict__ scl, float* __restrict__ out) {
    __shared__ _Float16 Xs[MTILE * XSTR];
    __shared__ _Float16 Hs[MTILE * HSTR];
    __shared__ int   stok[MTILE];
    __shared__ float ssc[MTILE];

    const int e = blockIdx.y;
    const int cnt = counts[e];
    const int row0 = blockIdx.x * MTILE;
    if (row0 >= cnt) return;

    const int tid = threadIdx.x;
    if (tid < MTILE) {
        int slot = row0 + tid;
        int ok = slot < cnt;
        stok[tid] = ok ? toks[e * NTOK + slot] : -1;
        ssc[tid]  = ok ? scl[e * NTOK + slot] : 0.f;
    }
    __syncthreads();

    // Stage scaled tokens as f16 (coalesced float2 reads).
    for (int p = tid; p < MTILE * (DIMD / 2); p += 256) {
        int r = p >> 9;
        int c = (p & 511) * 2;
        int t = stok[r];
        float sc = ssc[r];
        float2 v = make_float2(0.f, 0.f);
        if (t >= 0) v = *(const float2*)(x + (size_t)t * DIMD + c);
        Xs[r * XSTR + c]     = (_Float16)(v.x * sc);
        Xs[r * XSTR + c + 1] = (_Float16)(v.y * sc);
    }
    __syncthreads();

    const int wv   = tid >> 5;
    const int lane = tid & 31;
    const int half = lane >> 4;
    const int l15  = lane & 15;
    const int kb   = half * 8;    // A-frag K base
    const int kg   = half * 16;   // B-frag K group

    const WT* W1 = w1 + (size_t)e * DIMH * DIMD;
    const WT* W2 = w2 + (size_t)e * DIMD * DIMH;

    v8f acc[16];                  // 8 col-tiles x 2 m-tiles of persistent y
    v8f vzero = {0.f,0.f,0.f,0.f,0.f,0.f,0.f,0.f};
#pragma unroll
    for (int i = 0; i < 16; ++i) acc[i] = vzero;

    for (int hc = 0; hc < DIMH; hc += HCHUNK) {
        // ---- Phase A: Hs[:, 0..HCHUNK) = relu(Xs @ W1[hc..hc+HCHUNK,:]^T + b1)
#pragma unroll 1
        for (int ct = 0; ct < 8; ++ct) {
            int colL = wv * 128 + ct * 16 + l15;      // local h column
            int hrow = hc + colL;                     // w1 row / bias index
            const WT* wr = W1 + (size_t)hrow * DIMD + kg;
            v8f c0 = vzero, c1 = vzero;
#pragma unroll 4
            for (int kd = 0; kd < DIMD; kd += 32) {
                v16h bf = bfrag(wr + kd);
                v16h a0 = afrag_lds(&Xs[l15 * XSTR + kd], kb);
                v16h a1 = afrag_lds(&Xs[(16 + l15) * XSTR + kd], kb);
                c0 = wmma_f16(a0, bf, c0);
                c1 = wmma_f16(a1, bf, c1);
            }
            float bias = b1[(size_t)e * DIMH + hrow];
#pragma unroll
            for (int r = 0; r < 8; ++r) {
                float h0 = fmaxf(c0[r] + bias, 0.f);
                float h1 = fmaxf(c1[r] + bias, 0.f);
                Hs[(half * 8 + r) * HSTR + colL]      = (_Float16)h0;
                Hs[(16 + half * 8 + r) * HSTR + colL] = (_Float16)h1;
            }
        }
        __syncthreads();

        // ---- Phase B: y += Hs @ W2[:, hc..hc+HCHUNK]^T
#pragma unroll 1
        for (int ct = 0; ct < 8; ++ct) {
            int dcol = wv * 128 + ct * 16 + l15;      // output column
            const WT* wr = W2 + (size_t)dcol * DIMH + hc + kg;
            v8f y0 = acc[ct * 2], y1 = acc[ct * 2 + 1];
#pragma unroll 4
            for (int kh = 0; kh < HCHUNK; kh += 32) {
                v16h bf = bfrag(wr + kh);
                v16h a0 = afrag_lds(&Hs[l15 * HSTR + kh], kb);
                v16h a1 = afrag_lds(&Hs[(16 + l15) * HSTR + kh], kb);
                y0 = wmma_f16(a0, bf, y0);
                y1 = wmma_f16(a1, bf, y1);
            }
            acc[ct * 2] = y0; acc[ct * 2 + 1] = y1;
        }
        __syncthreads();
    }

    // ---- Epilogue: + b2, scatter rows to their tokens
#pragma unroll 1
    for (int ct = 0; ct < 8; ++ct) {
        int dcol = wv * 128 + ct * 16 + l15;
        float bias = b2[(size_t)e * DIMD + dcol];
        v8f y0 = acc[ct * 2], y1 = acc[ct * 2 + 1];
#pragma unroll
        for (int r = 0; r < 8; ++r) {
            int m0 = half * 8 + r;
            int t0 = stok[m0];
            if (t0 >= 0) out[(size_t)t0 * DIMD + dcol] = y0[r] + bias;
            int m1 = 16 + half * 8 + r;
            int t1 = stok[m1];
            if (t1 >= 0) out[(size_t)t1 * DIMD + dcol] = y1[r] + bias;
        }
    }
}

extern "C" void kernel_launch(void* const* d_in, const int* in_sizes, int n_in,
                              void* d_out, int out_size, void* d_ws, size_t ws_size,
                              hipStream_t stream) {
    const float* x  = (const float*)d_in[0];
    const float* gw = (const float*)d_in[1];
    const float* gb = (const float*)d_in[2];
    const float* w1 = (const float*)d_in[3];
    const float* b1 = (const float*)d_in[4];
    const float* w2 = (const float*)d_in[5];
    const float* b2 = (const float*)d_in[6];
    float* out = (float*)d_out;

    int*   counts = (int*)d_ws;                    // [8] (+pad to 64 ints)
    int*   toks   = counts + 64;                   // [8][8192]
    float* scl    = (float*)(toks + NEXP * NTOK);  // [8][8192]

    const size_t route_bytes = 64 * 4 + (size_t)NEXP * NTOK * 4 * 2;
    const size_t welems      = (size_t)NEXP * DIMH * DIMD;   // per weight tensor
    const size_t base        = (route_bytes + 255) & ~(size_t)255;
    const bool   f16_path    = ws_size >= base + 2 * welems * sizeof(_Float16);

    moe_zero_counts<<<1, 32, 0, stream>>>(counts);
    moe_gate<<<NTOK / 4, 128, 0, stream>>>(x, gw, gb, counts, toks, scl);

    dim3 grid(NTOK / MTILE, NEXP);                 // worst case: all tokens on one expert
    if (f16_path) {
        _Float16* w1h = (_Float16*)((char*)d_ws + base);
        _Float16* w2h = w1h + welems;
        moe_cvt_w<<<4096, 256, 0, stream>>>(w1, w1h, (long)(welems / 4));
        moe_cvt_w<<<4096, 256, 0, stream>>>(w2, w2h, (long)(welems / 4));
        moe_ffn<_Float16><<<grid, 256, 0, stream>>>(x, w1h, b1, w2h, b2,
                                                    counts, toks, scl, out);
    } else {
        moe_ffn<float><<<grid, 256, 0, stream>>>(x, w1, b1, w2, b2,
                                                 counts, toks, scl, out);
    }
}
